// EdgeResidualGraphBlock_65352222376579
// MI455X (gfx1250) — matrix operand order repaired and checked by reading, not compiled, not run on hardware
//
#include <hip/hip_runtime.h>
#include <hip/hip_bf16.h>

#define NNODES 10000
#define NEDGES 80000
#define HD     64
#define EDD    32
#define KB_TOT 130          // 4160 K / 32 per wmma step (4096 from W2 + 64 from b2)
#define KCHUNK 8            // kb steps staged in LDS per block iteration (nkb always even)
#define WAVES  4            // waves per block in fused kernel
#define HROW   65           // padded f32 row for h tile (64 h + 1.0 for bias fold)

typedef __attribute__((ext_vector_type(16))) __bf16 v16bf;
typedef __attribute__((ext_vector_type(8)))  float  v8f;

__device__ __forceinline__ float gelu_exact(float v) {
    return 0.5f * v * (1.0f + erff(v * 0.70710678118654752440f));
}

// ---------------- Kernel 1: h = gelu(edge_attr @ W1 + b1)  [E,64] ----------------
__global__ void edge_mlp1_kernel(const float* __restrict__ ea,
                                 const float* __restrict__ W1,
                                 const float* __restrict__ b1,
                                 float* __restrict__ hbuf) {
    int gid = blockIdx.x * 256 + threadIdx.x;
    if (gid >= NEDGES * HD) return;
    int e = gid >> 6, o = gid & 63;
    const float* row = ea + e * EDD;
    float acc = b1[o];
#pragma unroll
    for (int i = 0; i < EDD; ++i) acc = fmaf(row[i], W1[i * HD + o], acc);
    hbuf[gid] = gelu_exact(acc);
}

// ---- Kernel 2: swizzle W2 (+b2 as extra K rows) into WMMA B-fragment layout (bf16) ----
// Layout: frag[kb][nt][lane][j], j=0..15.  B 32x16 bf16 per ISA: lane half hi=lane>>4,
// per-lane K values are 8*hi + j (j<8), 16 + 8*hi + (j-8) (j>=8).  kidx = kb*32 + koff maps to (k,i):
// k = kb>>1, i = (kb&1)*32 + koff.  Element = W2[k, i*64 + n] (or b2[i*64+n] for k==64).
__global__ void build_w2frag_kernel(const float* __restrict__ W2,
                                    const float* __restrict__ b2,
                                    unsigned short* __restrict__ frag) {
    int idx = blockIdx.x * 256 + threadIdx.x;
    if (idx >= KB_TOT * 4 * 32 * 16) return;
    int j  = idx & 15;
    int L  = (idx >> 4) & 31;
    int nt = (idx >> 9) & 3;
    int kb = idx >> 11;
    int k  = kb >> 1;
    int i0 = (kb & 1) * 32;
    int hi = L >> 4;
    int n  = nt * 16 + (L & 15);
    int i  = i0 + 8 * hi + j + 8 * (j >> 3);
    float val = (k < HD) ? W2[k * (HD * HD) + i * HD + n] : b2[i * HD + n];
    __bf16 bv = (__bf16)val;
    unsigned short us;
    __builtin_memcpy(&us, &bv, 2);
    frag[idx] = us;
}

// ---------------- Kernel 3: fused  msg = P @ W2'  + scatter-add to agg ----------------
// Each wave owns a 16-edge tile.  Per lane: m = lane&15 (edge row), hi = lane>>4.
// The lane's entire x_src footprint is 32 f32 (runs 8*hi+16*r, r=0..3) -> registers.
__global__ void __launch_bounds__(WAVES * 32)
fused_msg_kernel(const float* __restrict__ x,
                 const int*   __restrict__ eidx,      // [2,E]: src then dst
                 const float* __restrict__ hbuf,      // [E,64]
                 const unsigned short* __restrict__ w2frag,
                 float* __restrict__ agg) {
    __shared__ float hl_lds[WAVES * 16 * HROW];                           // 16.6 KB
    __shared__ __align__(32) unsigned short b_lds[KCHUNK * 4 * 32 * 16];  // 32 KB

    const int tid  = threadIdx.x;
    const int wave = tid >> 5;
    const int lane = tid & 31;
    const int hi   = lane >> 4;
    const int m    = lane & 15;
    const int tile = blockIdx.x * WAVES + wave;      // 16-edge tile; 5000 total, exact

    const int* srcI = eidx;
    const int* dstI = eidx + NEDGES;

    float* hl = hl_lds + wave * 16 * HROW;

    // Gather h (f32) tile into LDS; fold bias row h[.][64]=1.
    for (int r = 0; r < 16; ++r) {
        int e = tile * 16 + r;
        hl[r * HROW + lane]      = hbuf[e * HD + lane];
        hl[r * HROW + 32 + lane] = hbuf[e * HD + 32 + lane];
    }
    if (lane < 16) hl[lane * HROW + 64] = 1.0f;

    // Registerize this lane's x_src window: xr[r*8+j] = x[src][8*hi + 16*r + j].
    float xr[32];
    {
        int s_m = srcI[tile * 16 + m];
        const float* xrow = x + (size_t)s_m * HD + 8 * hi;
#pragma unroll
        for (int r = 0; r < 4; ++r) {
            float4 q0 = ((const float4*)(xrow + 16 * r))[0];
            float4 q1 = ((const float4*)(xrow + 16 * r))[1];
            xr[r * 8 + 0] = q0.x; xr[r * 8 + 1] = q0.y;
            xr[r * 8 + 2] = q0.z; xr[r * 8 + 3] = q0.w;
            xr[r * 8 + 4] = q1.x; xr[r * 8 + 5] = q1.y;
            xr[r * 8 + 6] = q1.z; xr[r * 8 + 7] = q1.w;
        }
    }

    v8f acc[4];
#pragma unroll
    for (int nt = 0; nt < 4; ++nt)
#pragma unroll
        for (int q = 0; q < 8; ++q) acc[nt][q] = 0.0f;

    const v16bf* blv = (const v16bf*)b_lds;

    for (int c0 = 0; c0 < KB_TOT; c0 += KCHUNK) {
        int nkb = (KB_TOT - c0 < KCHUNK) ? (KB_TOT - c0) : KCHUNK;  // 8 or 2 (even)
        __syncthreads();
        {   // cooperative stage of B fragments: nkb*4096 bytes
            const uint4* gsrc = (const uint4*)w2frag + c0 * 256;
            uint4* ldst = (uint4*)b_lds;
            for (int t = tid; t < nkb * 256; t += WAVES * 32) ldst[t] = gsrc[t];
        }
        __syncthreads();

        int npair = nkb >> 1;
#pragma unroll 4
        for (int p = 0; p < npair; ++p) {
            float hk = hl[m * HROW + (c0 >> 1) + p];   // h[e,k], k = (c0 + 2p)/2

            {   // K-step 2p: i0 = 0 -> uses xr[0..15]
                v16bf a;
#pragma unroll
                for (int j = 0; j < 16; ++j) a[j] = (__bf16)(hk * xr[j]);
#pragma unroll
                for (int nt = 0; nt < 4; ++nt) {
                    v16bf b = blv[((2 * p) * 4 + nt) * 32 + lane];
                    acc[nt] = __builtin_amdgcn_wmma_f32_16x16x32_bf16(
                        false, a, false, b, (short)0, acc[nt], false, false);
                }
            }
            {   // K-step 2p+1: i0 = 32 -> uses xr[16..31]
                v16bf a;
#pragma unroll
                for (int j = 0; j < 16; ++j) a[j] = (__bf16)(hk * xr[16 + j]);
#pragma unroll
                for (int nt = 0; nt < 4; ++nt) {
                    v16bf b = blv[((2 * p + 1) * 4 + nt) * 32 + lane];
                    acc[nt] = __builtin_amdgcn_wmma_f32_16x16x32_bf16(
                        false, a, false, b, (short)0, acc[nt], false, false);
                }
            }
        }
    }

    // Scatter-add: C/D layout -> lane(L): n = nt*16 + (L&15); vgpr v -> m = v + 8*hi.
#pragma unroll
    for (int nt = 0; nt < 4; ++nt) {
#pragma unroll
        for (int v = 0; v < 8; ++v) {
            int mm = v + 8 * hi;
            int e  = tile * 16 + mm;
            int d  = dstI[e];
            unsafeAtomicAdd(&agg[d * HD + nt * 16 + (lane & 15)], acc[nt][v]);
        }
    }
}

// ---- Kernel 4: out = LN(gelu(agg + x@root + bias) + x) * gamma + beta ----
__global__ void finalize_kernel(const float* __restrict__ x,
                                const float* __restrict__ agg,
                                const float* __restrict__ root,
                                const float* __restrict__ bias,
                                const float* __restrict__ gamma,
                                const float* __restrict__ beta,
                                float* __restrict__ out) {
    __shared__ float rootl[HD * HD];   // 16 KB
    __shared__ float xrow[8][HD];
    int tid = threadIdx.x;
    for (int t = tid; t < HD * HD; t += 256) rootl[t] = root[t];
    int wave = tid >> 5, lane = tid & 31;
    int n = blockIdx.x * 8 + wave;     // 1250*8 == 10000, exact
    xrow[wave][lane]      = x[n * HD + lane];
    xrow[wave][lane + 32] = x[n * HD + 32 + lane];
    __syncthreads();

    int o0 = lane, o1 = lane + 32;
    float a0 = 0.0f, a1 = 0.0f;
#pragma unroll 8
    for (int i = 0; i < HD; ++i) {
        float xi = xrow[wave][i];
        a0 = fmaf(xi, rootl[i * HD + o0], a0);
        a1 = fmaf(xi, rootl[i * HD + o1], a1);
    }
    float v0 = agg[n * HD + o0] + a0 + bias[o0];
    float v1 = agg[n * HD + o1] + a1 + bias[o1];
    float p0 = gelu_exact(v0) + xrow[wave][o0];
    float p1 = gelu_exact(v1) + xrow[wave][o1];

    float s1 = p0 + p1, s2 = p0 * p0 + p1 * p1;
#pragma unroll
    for (int msk = 16; msk; msk >>= 1) {
        s1 += __shfl_xor(s1, msk, 32);
        s2 += __shfl_xor(s2, msk, 32);
    }
    float mu  = s1 * (1.0f / 64.0f);
    float var = s2 * (1.0f / 64.0f) - mu * mu;
    float inv = rsqrtf(var + 1e-5f);
    out[n * HD + o0] = (p0 - mu) * inv * gamma[o0] + beta[o0];
    out[n * HD + o1] = (p1 - mu) * inv * gamma[o1] + beta[o1];
}

extern "C" void kernel_launch(void* const* d_in, const int* in_sizes, int n_in,
                              void* d_out, int out_size, void* d_ws, size_t ws_size,
                              hipStream_t stream) {
    const float* x    = (const float*)d_in[0];
    const int*   eidx = (const int*)  d_in[1];   // [2,E]
    const float* ea   = (const float*)d_in[2];
    const float* W1   = (const float*)d_in[3];
    const float* b1   = (const float*)d_in[4];
    const float* W2   = (const float*)d_in[5];
    const float* b2   = (const float*)d_in[6];
    const float* root = (const float*)d_in[7];
    const float* bias = (const float*)d_in[8];
    const float* gam  = (const float*)d_in[9];
    const float* bet  = (const float*)d_in[10];
    float* out = (float*)d_out;

    // Workspace layout
    char* ws = (char*)d_ws;
    float*          hbuf   = (float*)ws;                                  // E*64 f32 = 20,480,000 B
    unsigned short* w2frag = (unsigned short*)(ws + 20480000);            // 130*4*32*16 bf16 = 532,480 B
    float*          agg    = (float*)(ws + 20480000 + 532480);            // N*64 f32 = 2,560,000 B

    hipMemsetAsync(agg, 0, (size_t)NNODES * HD * sizeof(float), stream);

    {   // h = gelu(edge_attr @ W1 + b1)
        int total = NEDGES * HD;
        edge_mlp1_kernel<<<(total + 255) / 256, 256, 0, stream>>>(ea, W1, b1, hbuf);
    }
    {   // swizzle W2/b2 -> B fragments
        int total = KB_TOT * 4 * 32 * 16;
        build_w2frag_kernel<<<(total + 255) / 256, 256, 0, stream>>>(W2, b2, w2frag);
    }
    {   // fused P @ W2' with scatter-add (5000 edge tiles / 4 waves per block)
        fused_msg_kernel<<<(NEDGES / 16) / WAVES, WAVES * 32, 0, stream>>>(
            x, eidx, hbuf, w2frag, agg);
    }
    {   // root GEMM + gelu + residual + LayerNorm
        finalize_kernel<<<NNODES / 8, 256, 0, stream>>>(x, agg, root, bias, gam, bet, out);
    }
}
